// MoELayer_87093346828353
// MI455X (gfx1250) — matrix operand order repaired and checked by reading, not compiled, not run on hardware
//
#include <hip/hip_runtime.h>
#include <hip/hip_bf16.h>

// ---------------------------------------------------------------------------
// MoE layer for MI455X (gfx1250, wave32, WMMA).
//  - Sparse top-2 routing (275 GFLOP instead of dense 1.1 TFLOP)
//  - bf16 inputs / fp32 accumulation via v_wmma_f32_16x16x32_bf16
//  - routed gather A-tiles loaded via TDM tensor_load_to_lds (gather mode)
//  - dense A-tiles staged to LDS via global_load_async_to_lds_b128 (ASYNCcnt)
//  - 4 N-tiles register-blocked per wave (A-fragment reuse, 4 indep acc chains)
// ---------------------------------------------------------------------------

#define B_TOK 8192
#define D_IN  1024
#define H_DIM 2048
#define O_DIM 1024
#define NE    8
#define NSLOT (B_TOK * 2)

typedef __bf16        v16bf __attribute__((ext_vector_type(16)));
typedef float         v8f   __attribute__((ext_vector_type(8)));
typedef unsigned int  v4u   __attribute__((ext_vector_type(4)));
typedef unsigned int  u32x4 __attribute__((ext_vector_type(4)));
typedef int           i32x4 __attribute__((ext_vector_type(4)));
typedef int           i32x8 __attribute__((ext_vector_type(8)));

union FragAB { v16bf v; v4u q[2]; };

// ---------------------------------------------------------------- converters
__global__ void cvt_bf16_kernel(const float* __restrict__ s,
                                __hip_bfloat16* __restrict__ d, long n) {
  long i = (long)blockIdx.x * blockDim.x + threadIdx.x;
  long st = (long)gridDim.x * blockDim.x;
  for (; i < n; i += st) d[i] = __float2bfloat16(s[i]);
}

// src: batch of [R,C] fp32 (row major) -> dst: batch of [C,R] bf16 (transposed)
__global__ void cvt_t_bf16_kernel(const float* __restrict__ s,
                                  __hip_bfloat16* __restrict__ d, int R, int C) {
  long nb = (long)R * C;
  const float* sb = s + (long)blockIdx.z * nb;
  __hip_bfloat16* db = d + (long)blockIdx.z * nb;
  long i = (long)blockIdx.x * blockDim.x + threadIdx.x;
  long st = (long)gridDim.x * blockDim.x;
  for (; i < nb; i += st) {
    int r = (int)(i / C), c = (int)(i - (long)r * C);
    db[(long)c * R + r] = __float2bfloat16(sb[i]);
  }
}

// ------------------------------------------------------------ unified GEMM
// C[row, 0..N) = act(A[arow(row), 0..K) @ W^T + bias[e])   (16-token M tiles)
// dense mode  (counts==nullptr): rows are 0..M_dense-1
// routed mode (counts!=nullptr): block (e,t) handles slots offsets[e]+t*16 ..,
//   A row index = token_of_slot[slot] (gather via TDM) or slot itself.
// flags: bit0 = relu, bit1 = fp32 output (else bf16)
__global__ __launch_bounds__(128)
void gemm_tile_kernel(const __hip_bfloat16* __restrict__ A,
                      const __hip_bfloat16* __restrict__ WT,   // [e][N][K] bf16
                      const float* __restrict__ bias,          // [e][N]
                      void* __restrict__ Cout,
                      const int* __restrict__ token_of_slot,
                      const int* __restrict__ counts,
                      const int* __restrict__ offsets,
                      int M_dense, int K, int N, int tiles_per_e, int flags) {
  __shared__ __align__(16) __hip_bfloat16 atile[16 * 2048];   // 64 KB max

  const int e = blockIdx.x / tiles_per_e;
  const int t = blockIdx.x - e * tiles_per_e;
  int rowbase, rem;
  if (counts) {
    const int cnt = counts[e];
    if (t * 16 >= cnt) return;
    rowbase = offsets[e] + t * 16;
    rem = cnt - t * 16;
  } else {
    rowbase = t * 16;
    rem = M_dense - rowbase;
    if (rem <= 0) return;
  }
  if (rem > 16) rem = 16;

  const __hip_bfloat16* wte = WT + (size_t)e * N * K;
  const float* be = bias + (size_t)e * N;
  const unsigned lds_base = (unsigned)(size_t)atile;   // low 32b = LDS offset

  if (token_of_slot) {
    // ---- TDM gather-mode load: 16 scattered token rows -> LDS, one D# ----
    if (threadIdx.x < 32) {                   // one wave issues the DMA
      int idxs[16];
#pragma unroll
      for (int r = 0; r < 16; ++r)
        idxs[r] = token_of_slot[rowbase + ((r < rem) ? r : 0)];
      const unsigned long long ga = (unsigned long long)(size_t)A;
      u32x4 g0;
      g0[0] = 0x80000001u;                    // count=1, gather_mode=1, 16b idx
      g0[1] = lds_base;                       // lds_addr
      g0[2] = (unsigned)(ga & 0xffffffffu);   // global_addr[31:0]
      g0[3] = (unsigned)((ga >> 32) & 0x01ffffffu) | (2u << 30);  // addr hi | type=2
      i32x8 g1;
      g1[0] = 1 << 16;                        // data_size = 2 bytes
      g1[1] = (K & 0xffff) << 16;             // tensor_dim0 lo16 (row length K)
      g1[2] = (B_TOK & 0xffff) << 16;         // dim0 hi(0) | tensor_dim1 lo16
      g1[3] = (K & 0xffff) << 16;             // dim1 hi(0) | tile_dim0 = K
      g1[4] = 16;                             // tile_dim1 = #valid indices
      g1[5] = K;                              // tensor_dim0_stride lo32
      g1[6] = 0;
      g1[7] = 0;
      i32x4 g2, g3;
#pragma unroll
      for (int j = 0; j < 4; ++j) {
        g2[j] = (idxs[2 * j] & 0xffff) | (idxs[2 * j + 1] << 16);
        g3[j] = (idxs[8 + 2 * j] & 0xffff) | (idxs[8 + 2 * j + 1] << 16);
      }
      i32x8 g4 = {0, 0, 0, 0, 0, 0, 0, 0};    // unused trailing group
      __builtin_amdgcn_tensor_load_to_lds(g0, g1, g2, g3, g4, 0);
      __builtin_amdgcn_s_wait_tensorcnt(0);
    }
  } else {
    // ---- per-lane async global->LDS copies (contiguous rows) ----
    const int chunks_per_row = K >> 3;        // 16B (8 bf16) chunks per row
    const int total_chunks = chunks_per_row << 4;
    for (int c = threadIdx.x; c < total_chunks; c += blockDim.x) {
      const int r   = c / chunks_per_row;
      const int col = c - r * chunks_per_row;
      const int ar  = (r < rem) ? r : 0;
      const unsigned long long src =
          (unsigned long long)(size_t)(A + (size_t)(rowbase + ar) * K) +
          (size_t)col * 16u;
      const unsigned lds = lds_base + (unsigned)(r * K + col * 8) * 2u;
      asm volatile("global_load_async_to_lds_b128 %0, %1, off"
                   :: "v"(lds), "v"(src) : "memory");
    }
    asm volatile("s_wait_asynccnt 0x0" ::: "memory");
  }
  __syncthreads();

  const int warp = threadIdx.x >> 5;
  const int lane = threadIdx.x & 31;
  const int mrow = lane & 15;         // A row within tile
  const int kh   = lane >> 4;         // K-half selector
  const int ngroups = N >> 6;         // groups of 4 N-tiles (64 columns)
  const __hip_bfloat16* arow_lds = atile + (size_t)mrow * K;

  for (int tg = warp; tg < ngroups; tg += 4) {
    const int n0 = (tg << 6) + (lane & 15);
    const __hip_bfloat16* brow = wte + (size_t)n0 * K;
    v8f acc[4];
#pragma unroll
    for (int j = 0; j < 4; ++j) {
      v8f z = {0.f, 0.f, 0.f, 0.f, 0.f, 0.f, 0.f, 0.f};
      acc[j] = z;
    }
    for (int kb = 0; kb < K; kb += 32) {
      FragAB fa;
      // A 16x32 bf16: lane half kh holds k = {kb+kh*8..+7} U {kb+16+kh*8..+7}
      fa.q[0] = *(const v4u*)((const char*)arow_lds + (size_t)(kb + kh * 8) * 2);
      fa.q[1] = *(const v4u*)((const char*)arow_lds + (size_t)(kb + 16 + kh * 8) * 2);
#pragma unroll
      for (int j = 0; j < 4; ++j) {
        // B 32x16 bf16: lane n, k = {kb+kh*8..+7} U {kb+16+kh*8..+7} in W^T row
        const char* bp = (const char*)(brow + (size_t)(j * 16) * K);
        FragAB fb;
        fb.q[0] = *(const v4u*)(bp + (size_t)(kb + kh * 8) * 2);
        fb.q[1] = *(const v4u*)(bp + (size_t)(kb + 16 + kh * 8) * 2);
        acc[j] = __builtin_amdgcn_wmma_f32_16x16x32_bf16(
            false, fa.v, false, fb.v, (short)0, acc[j], false, false);
      }
    }
#pragma unroll
    for (int j = 0; j < 4; ++j) {
      const int n = n0 + j * 16;
      const float bv = be[n];
      if (flags & 2) {
        float* Cf = (float*)Cout;
#pragma unroll
        for (int r = 0; r < 8; ++r) {
          const int mm = r + kh * 8;               // C/D layout: M = r + 8*kh
          if (mm < rem) {
            float v = acc[j][r] + bv;
            if (flags & 1) v = fmaxf(v, 0.f);
            Cf[(size_t)(rowbase + mm) * N + n] = v;
          }
        }
      } else {
        __hip_bfloat16* Cb = (__hip_bfloat16*)Cout;
#pragma unroll
        for (int r = 0; r < 8; ++r) {
          const int mm = r + kh * 8;
          if (mm < rem) {
            float v = acc[j][r] + bv;
            if (flags & 1) v = fmaxf(v, 0.f);
            Cb[(size_t)(rowbase + mm) * N + n] = __float2bfloat16(v);
          }
        }
      }
    }
  }
}

// --------------------------------------------------- gate stage 3 + top-2
__global__ void gate3_topk_kernel(const __hip_bfloat16* __restrict__ g2,
                                  const float* __restrict__ gw3,
                                  const float* __restrict__ gb3,
                                  float* __restrict__ probs_out,
                                  int* __restrict__ tk_idx,
                                  float* __restrict__ tk_w,
                                  int* __restrict__ counts) {
  int b = blockIdx.x * blockDim.x + threadIdx.x;
  if (b >= B_TOK) return;
  float lg[NE];
#pragma unroll
  for (int e = 0; e < NE; ++e) lg[e] = gb3[e];
  const __hip_bfloat16* row = g2 + (long)b * 128;
  for (int k = 0; k < 128; ++k) {
    float xv = __bfloat162float(row[k]);
#pragma unroll
    for (int e = 0; e < NE; ++e) lg[e] = fmaf(xv, gw3[k * NE + e], lg[e]);
  }
  float mx = lg[0];
  for (int e = 1; e < NE; ++e) mx = fmaxf(mx, lg[e]);
  float p[NE], s = 0.f;
  for (int e = 0; e < NE; ++e) { p[e] = expf(lg[e] - mx); s += p[e]; }
  float inv = 1.f / s;
  for (int e = 0; e < NE; ++e) {
    p[e] *= inv;
    probs_out[(long)b * NE + e] = p[e];
  }
  int i0 = 0;
  for (int e = 1; e < NE; ++e) if (p[e] > p[i0]) i0 = e;
  int i1 = (i0 == 0) ? 1 : 0;
  for (int e = 0; e < NE; ++e) if (e != i0 && p[e] > p[i1]) i1 = e;
  float den = p[i0] + p[i1];
  tk_idx[2 * b] = i0; tk_idx[2 * b + 1] = i1;
  tk_w[2 * b] = p[i0] / den; tk_w[2 * b + 1] = p[i1] / den;
  atomicAdd(&counts[i0], 1);
  atomicAdd(&counts[i1], 1);
}

// ------------------------------------------- usage (deterministic tree sum)
__global__ void usage_kernel(const float* __restrict__ probs,
                             float* __restrict__ usage) {
  __shared__ float red[256];
  const int e = blockIdx.x, t = threadIdx.x;
  float s = 0.f;
  for (int b = t; b < B_TOK; b += 256) s += probs[(long)b * NE + e];
  red[t] = s;
  __syncthreads();
  for (int w = 128; w > 0; w >>= 1) {
    if (t < w) red[t] += red[t + w];
    __syncthreads();
  }
  if (t == 0) usage[e] = red[0] / (float)B_TOK;
}

// ------------------------------- lb loss + routing offsets (single thread)
__global__ void lb_scan_kernel(const float* __restrict__ usage,
                               float* __restrict__ lb_out,
                               const int* __restrict__ counts,
                               int* __restrict__ offsets,
                               int* __restrict__ cursors) {
  if (threadIdx.x != 0 || blockIdx.x != 0) return;
  const float uni = 1.0f / (float)NE;
  float mean = 0.f;
  for (int e = 0; e < NE; ++e) mean += usage[e];
  mean /= (float)NE;
  float var = 0.f, kl = 0.f, ent = 0.f;
  for (int e = 0; e < NE; ++e) {
    float u = usage[e];
    var += (u - mean) * (u - mean);
    kl  += uni * (logf(uni) - logf(u + 1e-8f));
    ent -= u * logf(u + 1e-8f);
  }
  var /= (float)(NE - 1);
  const float max_ent = -logf(uni) * (float)NE;
  *lb_out = 0.01f * (0.5f * kl + 0.3f * var + 0.2f * (max_ent - ent));
  int off = 0;
  for (int e = 0; e < NE; ++e) { offsets[e] = off; off += counts[e]; cursors[e] = 0; }
}

__global__ void init_kernel(int* __restrict__ counts) {
  if (threadIdx.x < NE && blockIdx.x == 0) counts[threadIdx.x] = 0;
}

// -------------------------------------------------------------- routing
__global__ void route_kernel(const int* __restrict__ tk_idx,
                             const int* __restrict__ offsets,
                             int* __restrict__ cursors,
                             int* __restrict__ token_of_slot,
                             int* __restrict__ slot_of_token) {
  int b = blockIdx.x * blockDim.x + threadIdx.x;
  if (b >= B_TOK) return;
  for (int k = 0; k < 2; ++k) {
    int e = tk_idx[2 * b + k];
    int pos = atomicAdd(&cursors[e], 1);
    int s = offsets[e] + pos;
    token_of_slot[s] = b;
    slot_of_token[2 * b + k] = s;
  }
}

// -------------------------------------------------------------- combine
__global__ void combine_kernel(const float* __restrict__ out_buf,
                               const int* __restrict__ slot_of_token,
                               const float* __restrict__ tk_w,
                               float* __restrict__ out) {
  long i = (long)blockIdx.x * blockDim.x + threadIdx.x;
  if (i >= (long)B_TOK * O_DIM) return;
  int b = (int)(i >> 10);
  int o = (int)(i & 1023);
  int s0 = slot_of_token[2 * b], s1 = slot_of_token[2 * b + 1];
  out[i] = tk_w[2 * b] * out_buf[(size_t)s0 * O_DIM + o] +
           tk_w[2 * b + 1] * out_buf[(size_t)s1 * O_DIM + o];
}

// ---------------------------------------------------------------------------
extern "C" void kernel_launch(void* const* d_in, const int* in_sizes, int n_in,
                              void* d_out, int out_size, void* d_ws, size_t ws_size,
                              hipStream_t stream) {
  const float* x   = (const float*)d_in[0];
  const float* gw1 = (const float*)d_in[1];
  const float* gb1 = (const float*)d_in[2];
  const float* gw2 = (const float*)d_in[3];
  const float* gb2 = (const float*)d_in[4];
  const float* gw3 = (const float*)d_in[5];
  const float* gb3 = (const float*)d_in[6];
  const float* ew1 = (const float*)d_in[7];
  const float* eb1 = (const float*)d_in[8];
  const float* ew2 = (const float*)d_in[9];
  const float* eb2 = (const float*)d_in[10];
  const float* ew3 = (const float*)d_in[11];
  const float* eb3 = (const float*)d_in[12];

  float* out_final = (float*)d_out;                       // [B, O]
  float* out_lb    = out_final + (long)B_TOK * O_DIM;     // scalar
  float* out_probs = out_lb + 1;                          // [B, E]

  char* p = (char*)d_ws;
  auto carve = [&](size_t bytes) -> char* {
    char* r = p;
    p += (bytes + 255) & ~(size_t)255;
    return r;
  };
  __hip_bfloat16* xbf  = (__hip_bfloat16*)carve((size_t)B_TOK * D_IN * 2);
  __hip_bfloat16* g1   = (__hip_bfloat16*)carve((size_t)B_TOK * 256 * 2);
  __hip_bfloat16* g2   = (__hip_bfloat16*)carve((size_t)B_TOK * 128 * 2);
  __hip_bfloat16* gw1t = (__hip_bfloat16*)carve((size_t)256 * D_IN * 2);
  __hip_bfloat16* gw2t = (__hip_bfloat16*)carve((size_t)128 * 256 * 2);
  __hip_bfloat16* w1t  = (__hip_bfloat16*)carve((size_t)NE * H_DIM * D_IN * 2);
  __hip_bfloat16* w2t  = (__hip_bfloat16*)carve((size_t)NE * H_DIM * H_DIM * 2);
  __hip_bfloat16* w3t  = (__hip_bfloat16*)carve((size_t)NE * O_DIM * H_DIM * 2);
  __hip_bfloat16* h1   = (__hip_bfloat16*)carve((size_t)NSLOT * H_DIM * 2);
  __hip_bfloat16* h2   = (__hip_bfloat16*)carve((size_t)NSLOT * H_DIM * 2);
  float* out_buf       = (float*)carve((size_t)NSLOT * O_DIM * 4);
  int*   tk_idx        = (int*)carve((size_t)NSLOT * 4);
  float* tk_w          = (float*)carve((size_t)NSLOT * 4);
  int*   slot_of_token = (int*)carve((size_t)NSLOT * 4);
  int*   token_of_slot = (int*)carve((size_t)NSLOT * 4);
  int*   counts        = (int*)carve(64);
  int*   offsets       = (int*)carve(64);
  int*   cursors       = (int*)carve(64);
  float* usage         = (float*)carve(64);

  init_kernel<<<1, 32, 0, stream>>>(counts);

  // fp32 -> bf16 conversions (weights transposed to [N,K] for WMMA B-frags)
  cvt_bf16_kernel<<<2048, 256, 0, stream>>>(x, xbf, (long)B_TOK * D_IN);
  cvt_t_bf16_kernel<<<dim3(512, 1, 1), 256, 0, stream>>>(gw1, gw1t, D_IN, 256);
  cvt_t_bf16_kernel<<<dim3(128, 1, 1), 256, 0, stream>>>(gw2, gw2t, 256, 128);
  cvt_t_bf16_kernel<<<dim3(2048, 1, NE), 256, 0, stream>>>(ew1, w1t, D_IN, H_DIM);
  cvt_t_bf16_kernel<<<dim3(4096, 1, NE), 256, 0, stream>>>(ew2, w2t, H_DIM, H_DIM);
  cvt_t_bf16_kernel<<<dim3(2048, 1, NE), 256, 0, stream>>>(ew3, w3t, H_DIM, O_DIM);

  // gating MLP (dense WMMA GEMMs, relu, bf16 activations)
  gemm_tile_kernel<<<B_TOK / 16, 128, 0, stream>>>(
      xbf, gw1t, gb1, g1, nullptr, nullptr, nullptr,
      B_TOK, D_IN, 256, B_TOK / 16, /*relu*/ 1);
  gemm_tile_kernel<<<B_TOK / 16, 128, 0, stream>>>(
      g1, gw2t, gb2, g2, nullptr, nullptr, nullptr,
      B_TOK, 256, 128, B_TOK / 16, 1);

  // logits + softmax + top-2 + expert counts
  gate3_topk_kernel<<<B_TOK / 256, 256, 0, stream>>>(
      g2, gw3, gb3, out_probs, tk_idx, tk_w, counts);

  // load-balance loss + routing offsets
  usage_kernel<<<NE, 256, 0, stream>>>(out_probs, usage);
  lb_scan_kernel<<<1, 32, 0, stream>>>(usage, out_lb, counts, offsets, cursors);
  route_kernel<<<B_TOK / 256, 256, 0, stream>>>(
      tk_idx, offsets, cursors, token_of_slot, slot_of_token);

  // routed expert MLP: x -> h1 -> h2 -> out_buf  (bf16 WMMA, f32 accum)
  const int TPE = B_TOK / 16;   // worst case: one expert gets every token
  gemm_tile_kernel<<<NE * TPE, 128, 0, stream>>>(
      xbf, w1t, eb1, h1, token_of_slot, counts, offsets,
      0, D_IN, H_DIM, TPE, 1);
  gemm_tile_kernel<<<NE * TPE, 128, 0, stream>>>(
      h1, w2t, eb2, h2, nullptr, counts, offsets,
      0, H_DIM, H_DIM, TPE, 1);
  gemm_tile_kernel<<<NE * TPE, 128, 0, stream>>>(
      h2, w3t, eb3, out_buf, nullptr, counts, offsets,
      0, H_DIM, O_DIM, TPE, /*fp32 out, no relu*/ 2);

  // weighted combine of each token's two expert outputs
  combine_kernel<<<(B_TOK * O_DIM) / 256, 256, 0, stream>>>(
      out_buf, slot_of_token, tk_w, out_final);

  (void)in_sizes; (void)n_in; (void)out_size; (void)ws_size;
}
